// GRNN_20985210208331
// MI455X (gfx1250) — matrix operand order repaired and checked by reading, not compiled, not run on hardware
//
#include <hip/hip_runtime.h>
#include <hip/hip_bf16.h>

// MI455X (gfx1250) fused GRNN / Nadaraya-Watson kernel, v3.
//
// Roofline: 2 * N^2 * D * 2 FLOP ~= 69 GFLOP of bf16 WMMA vs operand
// traffic. The 32-row j-tile (row-major + transposed, 32KB) is staged in
// LDS once per 4-wave workgroup (4x less L2 traffic than per-wave fetch);
// j-range split across 2 WGs per row-block (256 WGs / 1024 waves).
// v3: staging now uses gfx1250 GLOBAL_LOAD_ASYNC_TO_LDS_B128 (ASYNCcnt,
// no VGPR round trip) with a triple-buffered pipeline: next tile's async
// loads are in flight while the current tile is consumed by WMMA, and only
// one barrier per iteration is needed (buffer reuse distance = 3).
// max-logit is exactly 0 -> no online softmax rescaling needed.

#define N_ROWS 8192
#define D_DIM  256
#define O_DIM  128
#define JSPLIT 2
#define JRANGE (N_ROWS / JSPLIT)
#define NIT    (JRANGE / 32)

typedef __attribute__((ext_vector_type(16))) __bf16 v16bf;
typedef __attribute__((ext_vector_type(8)))  __bf16 v8bf;
typedef __attribute__((ext_vector_type(8)))  float  v8f;
typedef __attribute__((ext_vector_type(4)))  float  v4f;

union Frag { v16bf v; v8bf h[2]; };

__device__ __forceinline__ v8bf ld8(const __bf16* p) { return *(const v8bf*)p; }

// One lane-wide async 16B global->LDS copy (ASYNCcnt-tracked, per-lane addrs).
__device__ __forceinline__ void async_ld16(unsigned lds_addr, const __bf16* g) {
  asm volatile("global_load_async_to_lds_b128 %0, %1, off"
               :: "v"(lds_addr), "v"(g) : "memory");
}

// ---------------------------------------------------------------------------
// Prep: sqh[i] = -||x_i||^2/2, xb = bf16(x) row-major, xbT = bf16(x)^T,
// Wb = bf16(W). One wave per row; 8 waves per block.
// ---------------------------------------------------------------------------
__global__ __launch_bounds__(256) void grnn_prep(const float* __restrict__ x,
                                                 const float* __restrict__ W,
                                                 __bf16* __restrict__ xb,
                                                 __bf16* __restrict__ xbT,
                                                 __bf16* __restrict__ Wb,
                                                 float*  __restrict__ sqh) {
  const int lane = threadIdx.x & 31;
  const int wave = threadIdx.x >> 5;
  const int row  = blockIdx.x * 8 + wave;
  if (row < N_ROWS) {
    float s = 0.f;
#pragma unroll
    for (int k = 0; k < 8; ++k) {
      const int d = lane + k * 32;
      const float v = x[row * D_DIM + d];
      s += v * v;
      const __bf16 bv = (__bf16)v;
      xb [row * D_DIM + d] = bv;
      xbT[d * N_ROWS + row] = bv;
    }
#pragma unroll
    for (int off = 16; off > 0; off >>= 1) s += __shfl_xor(s, off, 32);
    if (lane == 0) sqh[row] = -0.5f * s;
  } else {
    const int wrow = row - N_ROWS;
    if (wrow < O_DIM) {
#pragma unroll
      for (int k = 0; k < 8; ++k) {
        const int d = lane + k * 32;
        Wb[wrow * D_DIM + d] = (__bf16)W[wrow * D_DIM + d];
      }
    }
  }
}

// ---------------------------------------------------------------------------
// Main fused kernel. Grid: (8192/64 row-blocks) x (2 j-halves) = 256 WGs of
// 4 waves. Each wave owns 16 rows; the WG cooperatively async-stages the
// 32-row x_j tile into LDS (row-major XJ for S^T A-frags, transposed XJT
// for P@X B-frags), triple-buffered so global latency overlaps WMMA.
// Partial (num, den) per j-half go to private f32 buffers (no atomics).
// ---------------------------------------------------------------------------
__global__ __launch_bounds__(128) void grnn_main(const __bf16* __restrict__ xb,
                                                 const __bf16* __restrict__ xbT,
                                                 const float*  __restrict__ sqh,
                                                 float* __restrict__ num,
                                                 float* __restrict__ den) {
  const int tid   = threadIdx.x;
  const int lane  = tid & 31;
  const int wave  = tid >> 5;
  const int hf    = lane >> 4;
  const int l16   = lane & 15;
  const int rowbk = blockIdx.x >> 1;
  const int jh    = blockIdx.x & 1;
  const int ibase = rowbk * 64 + wave * 16;

  float* __restrict__ numb = num + (size_t)jh * N_ROWS * D_DIM;
  float* __restrict__ denb = den + jh * N_ROWS;

  __shared__ __align__(16) __bf16 XJ [3][32 * D_DIM];   // 3 x 16KB j-tiles
  __shared__ __align__(16) __bf16 XJT[3][D_DIM * 32];   // 3 x 16KB transposed
  __shared__ __align__(16) __bf16 P  [4][16 * 32];      // 1KB per-wave P tile

  // Issue one tile's staging: 16 async b128 per thread (8 XJ + 8 XJT).
  auto stage = [&](int buf, int jb) {
    {
      const __bf16* g = xb + (jb + (tid >> 2)) * D_DIM + (tid & 3) * 64;
      unsigned l = (unsigned)(unsigned long long)
                   &XJ[buf][(tid >> 2) * D_DIM + (tid & 3) * 64];
#pragma unroll
      for (int k = 0; k < 8; ++k) async_ld16(l + k * 16, g + k * 8);
    }
#pragma unroll
    for (int r = 0; r < 2; ++r) {
      const int d = tid * 2 + r;
      const __bf16* g = xbT + (size_t)d * N_ROWS + jb;
      unsigned l = (unsigned)(unsigned long long)&XJT[buf][d * 32];
#pragma unroll
      for (int k = 0; k < 4; ++k) async_ld16(l + k * 16, g + k * 8);
    }
  };

  // Resident B fragments for S^T gemm: rows of x_i.
  // B-layout: lane l holds n=i=l16, K = c*32 + hf*8 + {0..7} and +16.
  Frag Bi[8];
#pragma unroll
  for (int c = 0; c < 8; ++c) {
    const __bf16* p = xb + (ibase + l16) * D_DIM + c * 32 + hf * 8;
    Bi[c].h[0] = ld8(p);
    Bi[c].h[1] = ld8(p + 16);
  }
  const float sqih = sqh[ibase + l16];

  v8f acc[16] = {};
  float dacc = 0.f;

  stage(0, jh * JRANGE);                  // prologue: tile 0 in flight

  for (int it = 0; it < NIT; ++it) {
    const int jb  = jh * JRANGE + it * 32;
    const int buf = it % 3;

    if (it + 1 < NIT) {
      stage((it + 1) % 3, jb + 32);       // next tile in flight (other buffer)
      asm volatile("s_wait_asynccnt 0x10" ::: "memory");  // tile `it` landed
    } else {
      asm volatile("s_wait_asynccnt 0x0" ::: "memory");
    }
    __syncthreads();                      // all waves see tile `it` in LDS

    // C init folds the logit bias: s = x_i.x_j - sq_i/2 - sq_j/2.
    // S^T C-layout: lane l holds i=l16, j = vgpr + 8*hf (+16 for tile 1).
    v8f s0, s1;
#pragma unroll
    for (int v = 0; v < 8; ++v) {
      s0[v] = sqih + sqh[jb + hf * 8 + v];
      s1[v] = sqih + sqh[jb + 16 + hf * 8 + v];
    }
#pragma unroll
    for (int c = 0; c < 8; ++c) {
      // A-layout: lane l holds m=j=l16, K = c*32 + hf*8 + {0..7} and +16.
      Frag a0, a1;
      const __bf16* p0 = &XJ[buf][l16 * D_DIM + c * 32 + hf * 8];
      a0.h[0] = *(const v8bf*)p0;
      a0.h[1] = *(const v8bf*)(p0 + 16);
      const __bf16* p1 = &XJ[buf][(16 + l16) * D_DIM + c * 32 + hf * 8];
      a1.h[0] = *(const v8bf*)p1;
      a1.h[1] = *(const v8bf*)(p1 + 16);
      s0 = __builtin_amdgcn_wmma_f32_16x16x32_bf16(false, a0.v, false, Bi[c].v,
                                                   (short)0, s0, false, false);
      s1 = __builtin_amdgcn_wmma_f32_16x16x32_bf16(false, a1.v, false, Bi[c].v,
                                                   (short)0, s1, false, false);
    }
    // p = exp(logit) in (0,1]; accumulate den; stash bf16 P tile (wave-private).
#pragma unroll
    for (int v = 0; v < 8; v += 2) {
      union { __bf16 h2[2]; unsigned u; } pk;
      const float p0 = __expf(s0[v]), p1 = __expf(s0[v + 1]);
      dacc += p0 + p1;
      pk.h2[0] = (__bf16)p0; pk.h2[1] = (__bf16)p1;
      *(unsigned*)&P[wave][l16 * 32 + hf * 8 + v] = pk.u;
      const float q0 = __expf(s1[v]), q1 = __expf(s1[v + 1]);
      dacc += q0 + q1;
      pk.h2[0] = (__bf16)q0; pk.h2[1] = (__bf16)q1;
      *(unsigned*)&P[wave][l16 * 32 + 16 + hf * 8 + v] = pk.u;
    }
    // Same-wave LDS store->load: HW keeps LDS in order; stop compiler reorder.
    asm volatile("" ::: "memory");
    Frag pa;           // P as A-frag: lane l: m=i=l16, K=j = hf*16 + 0..15 contiguous
    {
      const __bf16* pp = &P[wave][l16 * 32 + hf * 16];
      pa.h[0] = *(const v8bf*)pp;
      pa.h[1] = *(const v8bf*)(pp + 8);
    }
    // acc += P @ X_j : B-frags from XJT, lane l: n=d=l16 in tile,
    // K = hf*16 + 0..15 contiguous.
#pragma unroll
    for (int dt = 0; dt < 16; ++dt) {
      Frag bv;
      const __bf16* pv = &XJT[buf][(dt * 16 + l16) * 32 + hf * 16];
      bv.h[0] = *(const v8bf*)pv;
      bv.h[1] = *(const v8bf*)(pv + 8);
      acc[dt] = __builtin_amdgcn_wmma_f32_16x16x32_bf16(false, pa.v, false, bv.v,
                                                        (short)0, acc[dt],
                                                        false, false);
    }
  }

  // den lives at lane l for i=l16 in two half copies -> fold halves, store.
  dacc += __shfl_xor(dacc, 16, 32);
  if (lane < 16) denb[ibase + l16] = dacc;
  // acc C-layout: lane l holds i = vgpr + 8*hf, d = dt*16 + l16.
#pragma unroll
  for (int v = 0; v < 8; ++v) {
#pragma unroll
    for (int dt = 0; dt < 16; ++dt) {
      numb[(ibase + v + 8 * hf) * D_DIM + dt * 16 + l16] = acc[dt][v];
    }
  }
}

// ---------------------------------------------------------------------------
// Head: fold j-half partials, normalize, and out = H @ W^T + b via WMMA.
// One wave per 16-row i-tile (512 waves); bias in the C initializer.
// ---------------------------------------------------------------------------
__global__ __launch_bounds__(128) void grnn_head(const float* __restrict__ num,
                                                 const float* __restrict__ den,
                                                 const __bf16* __restrict__ Wb,
                                                 const float*  __restrict__ bias,
                                                 float* __restrict__ out) {
  const int lane  = threadIdx.x & 31;
  const int wave  = threadIdx.x >> 5;
  const int hf    = lane >> 4;
  const int l16   = lane & 15;
  const int ibase = (blockIdx.x * 4 + wave) * 16;

  const float* __restrict__ num0 = num;
  const float* __restrict__ num1 = num + (size_t)N_ROWS * D_DIM;

  // Per-lane row normalizer: A-frag lane l holds row i = ibase + l16.
  const float dtot = den[ibase + l16] + den[N_ROWS + ibase + l16];
  const float r = __builtin_amdgcn_rcpf(dtot);

  // Build resident A frags: h = (num0+num1)*r -> bf16 in A-layout
  // (lane l: m=i=l16, K = c*32 + hf*8 + {0..7} and +16).
  Frag A[8];
#pragma unroll
  for (int c = 0; c < 8; ++c) {
    const int base = (ibase + l16) * D_DIM + c * 32 + hf * 8;
#pragma unroll
    for (int piece = 0; piece < 2; ++piece) {          // lo (+0), hi (+16)
      const int off = base + piece * 16;
      const v4f a0 = *(const v4f*)(num0 + off);
      const v4f a1 = *(const v4f*)(num0 + off + 4);
      const v4f b0 = *(const v4f*)(num1 + off);
      const v4f b1 = *(const v4f*)(num1 + off + 4);
#pragma unroll
      for (int e = 0; e < 4; ++e) {
        A[c].v[piece * 8 + e]     = (__bf16)((a0[e] + b0[e]) * r);
        A[c].v[piece * 8 + 4 + e] = (__bf16)((a1[e] + b1[e]) * r);
      }
    }
  }
#pragma unroll
  for (int ot = 0; ot < 8; ++ot) {
    const float bo = bias[ot * 16 + l16];   // n=o=l16 in this tile
    v8f cacc;
#pragma unroll
    for (int v = 0; v < 8; ++v) cacc[v] = bo;
#pragma unroll
    for (int c = 0; c < 8; ++c) {
      Frag bw;  // B-frag: lane l: n=o, K = c*32 + hf*8 + {0..7} and +16 (rows of W)
      const __bf16* p = Wb + (ot * 16 + l16) * D_DIM + c * 32 + hf * 8;
      bw.h[0] = ld8(p);
      bw.h[1] = ld8(p + 16);
      cacc = __builtin_amdgcn_wmma_f32_16x16x32_bf16(false, A[c].v, false, bw.v,
                                                     (short)0, cacc, false, false);
    }
#pragma unroll
    for (int v = 0; v < 8; ++v)
      out[(ibase + v + 8 * hf) * O_DIM + ot * 16 + l16] = cacc[v];
  }
}

// ---------------------------------------------------------------------------
extern "C" void kernel_launch(void* const* d_in, const int* in_sizes, int n_in,
                              void* d_out, int out_size, void* d_ws, size_t ws_size,
                              hipStream_t stream) {
  const float* x = (const float*)d_in[0];   // [8192,256]
  const float* W = (const float*)d_in[1];   // [128,256]
  const float* b = (const float*)d_in[2];   // [128]

  char* ws = (char*)d_ws;
  __bf16* xb  = (__bf16*)(ws);                                 // 4 MiB
  __bf16* xbT = (__bf16*)(ws + (4u << 20));                    // 4 MiB
  float*  sqh = (float*) (ws + (8u << 20));                    // 32 KiB
  __bf16* Wb  = (__bf16*)(ws + (8u << 20) + (64u << 10));      // 64 KiB
  float*  num = (float*) (ws + (8u << 20) + (128u << 10));     // 16 MiB (2 halves)
  float*  den = (float*) (ws + (24u << 20) + (128u << 10));    // 64 KiB (2 halves)

  // rows 0..8191 -> x prep; rows 8192..8319 -> W conversion
  grnn_prep<<<(N_ROWS + O_DIM) / 8, 256, 0, stream>>>(x, W, xb, xbT, Wb, sqh);
  grnn_main<<<(N_ROWS / 64) * JSPLIT, 128, 0, stream>>>(xb, xbT, sqh, num, den);
  grnn_head<<<N_ROWS / 64, 128, 0, stream>>>(num, den, Wb, b, (float*)d_out);
}
